// BayesianKANLayer_7395933684400
// MI455X (gfx1250) — compile-verified, matrix operand order checked by reading, not run on hardware
//
#include <hip/hip_runtime.h>
#include <hip/hip_bf16.h>

// ---------------------------------------------------------------------------
// BayesianKANLayer for MI455X (gfx1250, wave32, WMMA, async-to-LDS)
//   out[b,o] = sum_{i,k} basis(x[b,i])[k] * (mean + eps*exp(0.5*lv))[o,i,k]
//   kl       = 0.5 * sum(exp(lv) + mean^2 - 1 - lv)
// Pipeline:
//   1) kan_basis      : f32 x -> f16 basis  A[4096][8192]      (ws +0,   64MB)
//   2) kan_coeffs_kl  : f16 coeffs Bt[512][8192] + KL partials (ws +64MB, 8MB / +72MB)
//   3) kan_kl_final   : tree-reduce partials -> d_out[M*N]
//   4) kan_gemm       : f16 WMMA GEMM, double-buffered GLOBAL_LOAD_ASYNC_TO_LDS
// ---------------------------------------------------------------------------

typedef __attribute__((ext_vector_type(16))) _Float16 v16h;
typedef __attribute__((ext_vector_type(8)))  _Float16 v8h;
typedef __attribute__((ext_vector_type(4)))  _Float16 v4h;
typedef __attribute__((ext_vector_type(8)))  float    v8f;

#define KAN_B     4096
#define KAN_IN    512
#define KAN_OUT   512
#define KAN_NB    16
#define KAN_K     (KAN_IN * KAN_NB)   // 8192
#define KAN_NELEM (KAN_OUT * KAN_IN * KAN_NB)

// LDS row padding: 64 data halves + 8 pad = 72 halves (144B = 36 dwords, 16B
// aligned). Row stride of 36 dwords puts the 16 fragment rows on 16 distinct
// 4-bank groups of the 64-bank LDS -> conflict-free ds_load_b128.
#define LDS_W 72

// ------------------------- 1) B-spline basis (f32 -> f16) -------------------
__global__ __launch_bounds__(256) void kan_basis(const float* __restrict__ x,
                                                 _Float16* __restrict__ basis) {
  int idx = blockIdx.x * blockDim.x + threadIdx.x;   // one (b,i) pair per thread
  float xv = x[idx];
  float xb = fminf(fmaxf(xv, -1.0f), 1.0f - 1e-6f);

  float knots[20];
#pragma unroll
  for (int i = 0; i < 20; ++i) {
    int j = (i < 3) ? 0 : ((i > 16) ? 13 : (i - 3));
    knots[i] = -1.0f + 2.0f * (float)j * (1.0f / 13.0f);
  }

  float Bv[19];
#pragma unroll
  for (int j = 0; j < 19; ++j)
    Bv[j] = (xb >= knots[j] && xb < knots[j + 1]) ? 1.0f : 0.0f;

  const float eps = 1e-8f;
#pragma unroll
  for (int d = 1; d <= 3; ++d) {
#pragma unroll
    for (int j = 0; j < 19 - 3; ++j) {
      if (j < 19 - d) {
        float left  = (xb - knots[j]) / (knots[j + d] - knots[j] + eps) * Bv[j];
        float right = (knots[j + d + 1] - xb) /
                      (knots[j + d + 1] - knots[j + 1] + eps) * Bv[j + 1];
        Bv[j] = left + right;
      }
    }
  }

  v8h lo, hi;
#pragma unroll
  for (int j = 0; j < 8; ++j) { lo[j] = (_Float16)Bv[j]; hi[j] = (_Float16)Bv[j + 8]; }
  v8h* out = (v8h*)(basis + (size_t)idx * KAN_NB);
  out[0] = lo;
  out[1] = hi;
}

// --------------- 2) coeffs (reparameterized) + KL block partials ------------
__global__ __launch_bounds__(256) void kan_coeffs_kl(const float* __restrict__ mean,
                                                     const float* __restrict__ lv,
                                                     const float* __restrict__ eps,
                                                     _Float16* __restrict__ coeffs,
                                                     float* __restrict__ partial) {
  __shared__ float red[256];
  int t = blockIdx.x * blockDim.x + threadIdx.x;
  int base = t * 4;

  float4 m = *(const float4*)(mean + base);
  float4 l = *(const float4*)(lv + base);
  float4 e = *(const float4*)(eps + base);

  float v0 = __expf(l.x), v1 = __expf(l.y), v2 = __expf(l.z), v3 = __expf(l.w);
  float s0 = __expf(0.5f * l.x), s1 = __expf(0.5f * l.y);
  float s2 = __expf(0.5f * l.z), s3 = __expf(0.5f * l.w);

  v4h c;
  c[0] = (_Float16)(m.x + e.x * s0);
  c[1] = (_Float16)(m.y + e.y * s1);
  c[2] = (_Float16)(m.z + e.z * s2);
  c[3] = (_Float16)(m.w + e.w * s3);
  *(v4h*)(coeffs + base) = c;

  float kl = 0.5f * ((v0 + m.x * m.x - 1.0f - l.x) +
                     (v1 + m.y * m.y - 1.0f - l.y) +
                     (v2 + m.z * m.z - 1.0f - l.z) +
                     (v3 + m.w * m.w - 1.0f - l.w));

  red[threadIdx.x] = kl;
  __syncthreads();
#pragma unroll
  for (int off = 128; off > 0; off >>= 1) {
    if (threadIdx.x < off) red[threadIdx.x] += red[threadIdx.x + off];
    __syncthreads();
  }
  if (threadIdx.x == 0) partial[blockIdx.x] = red[0];
}

// ----------------------- 3) KL finalize (deterministic) ---------------------
__global__ __launch_bounds__(256) void kan_kl_final(const float* __restrict__ partial,
                                                    float* __restrict__ out_kl, int n) {
  __shared__ float red[256];
  float acc = 0.0f;
  for (int i = threadIdx.x; i < n; i += 256) acc += partial[i];
  red[threadIdx.x] = acc;
  __syncthreads();
#pragma unroll
  for (int off = 128; off > 0; off >>= 1) {
    if (threadIdx.x < off) red[threadIdx.x] += red[threadIdx.x + off];
    __syncthreads();
  }
  if (threadIdx.x == 0) *out_kl = red[0];
}

// ---------------- async global -> LDS (ASYNCcnt-tracked, 16B) ---------------
__device__ __forceinline__ void kan_async_ld16(unsigned lds_addr, unsigned gvoff,
                                               const void* sbase) {
  // GVS mode: mem = SADDR(sgpr64) + VADDR[31:0]; VDST = per-lane LDS byte addr.
  asm volatile("global_load_async_to_lds_b128 %0, %1, %2"
               :: "v"(lds_addr), "v"(gvoff), "s"(sbase)
               : "memory");
}

// ------------------- 4) f16 WMMA GEMM: C = A(MxK) * Bt(NxK)^T ---------------
// 128x64 C tile / 256-thread block (8 wave32 waves; each wave 32x32).
// K tiled by 64, double-buffered via GLOBAL_LOAD_ASYNC_TO_LDS_B128.
__global__ __launch_bounds__(256) void kan_gemm(const _Float16* __restrict__ A,
                                                const _Float16* __restrict__ Bt,
                                                float* __restrict__ C) {
  const int K = KAN_K;        // 8192
  const int N = KAN_OUT;      // 512

  __shared__ __align__(16) _Float16 As[2][128][LDS_W];   // 36 KB
  __shared__ __align__(16) _Float16 Bs[2][64][LDS_W];    // 18 KB

  const int tid  = threadIdx.x;
  const int lane = tid & 31;
  const int wave = tid >> 5;
  const int wm   = wave & 3;          // 4 row groups of 32
  const int wn   = wave >> 2;         // 2 col groups of 32
  const int mBase = blockIdx.y * 128;
  const int nBase = blockIdx.x * 64;

  // 16-bit A/B fragment K layout (wave32): lanes 0-15 hold K {0..7,16..23},
  // lanes 16-31 hold K {8..15,24..31}.
  const int kho0 = (lane < 16) ? 0 : 8;
  const int kho1 = kho0 + 16;
  const int fr   = lane & 15;

  const v8f vzero = {0.f, 0.f, 0.f, 0.f, 0.f, 0.f, 0.f, 0.f};
  v8f acc[2][2];
#pragma unroll
  for (int mt = 0; mt < 2; ++mt)
#pragma unroll
    for (int nt = 0; nt < 2; ++nt) acc[mt][nt] = vzero;

  // Issue one K-tile's async loads: A 128x64 halves (1024 x 16B chunks, 4/thr)
  // + B 64x64 halves (512 chunks, 2/thr) => 6 async instructions per wave.
  auto issue = [&](int k0, int buf) {
#pragma unroll
    for (int j = 0; j < 4; ++j) {
      int c = tid + j * 256;
      int row = c >> 3, col = (c & 7) << 3;
      unsigned lds = (unsigned)(unsigned long long)&As[buf][row][col];
      unsigned off = (unsigned)((mBase + row) * K + k0 + col) * 2u;
      kan_async_ld16(lds, off, A);
    }
#pragma unroll
    for (int j = 0; j < 2; ++j) {
      int c = tid + j * 256;
      int row = c >> 3, col = (c & 7) << 3;
      unsigned lds = (unsigned)(unsigned long long)&Bs[buf][row][col];
      unsigned off = (unsigned)((nBase + row) * K + k0 + col) * 2u;
      kan_async_ld16(lds, off, Bt);
    }
  };

  auto compute = [&](int buf) {
#pragma unroll
    for (int kk = 0; kk < 64; kk += 32) {
      v16h af[2], bf[2];
#pragma unroll
      for (int mt = 0; mt < 2; ++mt) {
        const _Float16* p = &As[buf][wm * 32 + mt * 16 + fr][kk];
        v8h lo = *(const v8h*)(p + kho0);
        v8h hi = *(const v8h*)(p + kho1);
        af[mt] = __builtin_shufflevector(lo, hi, 0, 1, 2, 3, 4, 5, 6, 7,
                                         8, 9, 10, 11, 12, 13, 14, 15);
      }
#pragma unroll
      for (int nt = 0; nt < 2; ++nt) {
        const _Float16* p = &Bs[buf][wn * 32 + nt * 16 + fr][kk];
        v8h lo = *(const v8h*)(p + kho0);
        v8h hi = *(const v8h*)(p + kho1);
        bf[nt] = __builtin_shufflevector(lo, hi, 0, 1, 2, 3, 4, 5, 6, 7,
                                         8, 9, 10, 11, 12, 13, 14, 15);
      }
#pragma unroll
      for (int mt = 0; mt < 2; ++mt)
#pragma unroll
        for (int nt = 0; nt < 2; ++nt)
          acc[mt][nt] = __builtin_amdgcn_wmma_f32_16x16x32_f16(
              false, af[mt], false, bf[nt], (short)0, acc[mt][nt], false, false);
    }
  };

  const int T = K / 64;   // 128 K-tiles
  issue(0, 0);
  int t = 0;
  for (; t < T - 1; ++t) {
    issue((t + 1) * 64, (t + 1) & 1);                 // prefetch next tile
    asm volatile("s_wait_asynccnt 6" ::: "memory");   // tile t resident
    __syncthreads();
    compute(t & 1);
    __syncthreads();                                  // all reads done before overwrite
  }
  asm volatile("s_wait_asynccnt 0" ::: "memory");
  __syncthreads();
  compute(t & 1);

  // C/D layout (16x16 f32, 8 VGPRs): N = lane%16; M = v + 8*(lane>=16)
  const int mrow = (lane >> 4) << 3;
#pragma unroll
  for (int mt = 0; mt < 2; ++mt)
#pragma unroll
    for (int nt = 0; nt < 2; ++nt)
#pragma unroll
      for (int v = 0; v < 8; ++v) {
        int m = mBase + wm * 32 + mt * 16 + mrow + v;
        int n = nBase + wn * 32 + nt * 16 + fr;
        C[(size_t)m * N + n] = acc[mt][nt][v];
      }
}

// ----------------------------------------------------------------------------
extern "C" void kernel_launch(void* const* d_in, const int* in_sizes, int n_in,
                              void* d_out, int out_size, void* d_ws, size_t ws_size,
                              hipStream_t stream) {
  const float* x    = (const float*)d_in[0];
  const float* mean = (const float*)d_in[1];
  const float* lv   = (const float*)d_in[2];
  const float* eps  = (const float*)d_in[3];
  float* out = (float*)d_out;

  char* ws = (char*)d_ws;
  _Float16* basis   = (_Float16*)ws;                                  // 64 MB
  _Float16* coeffs  = (_Float16*)(ws + (size_t)64 * 1024 * 1024);     //  8 MB
  float*    partial = (float*)(ws + (size_t)72 * 1024 * 1024);        // 16 KB

  // 1) basis: 4096*512 scalars, one per thread
  kan_basis<<<(KAN_B * KAN_IN) / 256, 256, 0, stream>>>(x, basis);

  // 2) coeffs + KL partials: 4096 blocks x 256 threads x 4 elems
  kan_coeffs_kl<<<KAN_NELEM / (256 * 4), 256, 0, stream>>>(mean, lv, eps, coeffs, partial);

  // 3) KL finalize -> d_out[M*N]
  kan_kl_final<<<1, 256, 0, stream>>>(partial, out + (size_t)KAN_B * KAN_OUT,
                                      KAN_NELEM / (256 * 4));

  // 4) WMMA GEMM: grid (N/64, M/128) = (8, 32) = 256 blocks
  dim3 grid(KAN_OUT / 64, KAN_B / 128);
  kan_gemm<<<grid, 256, 0, stream>>>(basis, coeffs, out);
}
